// Indexer_25443386262317
// MI455X (gfx1250) — matrix operand order repaired and checked by reading, not compiled, not run on hardware
//
#include <hip/hip_runtime.h>
#include <hip/hip_bf16.h>
#include <stdint.h>

// Problem constants (from reference)
#define BB    2
#define SS    4096
#define DIMD  4096
#define QLRK  1536
#define NH    32
#define HD    128
#define RD    64
#define TOPKN 2048
#define BS    (BB * SS)

#define HSCALE  0.08838834764831845f   // 1/sqrt(128)
#define SMSCALE 0.08838834764831845f   // 1/sqrt(128)
#define NHSCALE 0.17677669529663689f   // 1/sqrt(32)

typedef __attribute__((ext_vector_type(16))) __bf16        v16bf;
typedef __attribute__((ext_vector_type(8)))  float         v8f;
typedef __attribute__((ext_vector_type(4)))  unsigned int  v4u;
typedef __attribute__((ext_vector_type(4)))  int           v4i;
typedef __attribute__((ext_vector_type(8)))  int           v8i;

__device__ __forceinline__ v8f wmma_bf16(v16bf a, v16bf b, v8f c) {
  return __builtin_amdgcn_wmma_f32_16x16x32_bf16(false, a, false, b,
                                                 (short)0, c, false, false);
}

// A-fragment (16x32 bf16): lane m = lane&15; lo-lanes K=k0..k0+15,
// hi-lanes +16. Source is row-major f32 (converted) or bf16.
__device__ __forceinline__ v16bf load_a_f32(const float* p0, int ld, int k0, int lane) {
  int m = lane & 15, hi = lane >> 4;
  const float* p = p0 + (size_t)m * ld + k0 + hi * 16;
  v16bf a;
#pragma unroll
  for (int i = 0; i < 8; ++i) a[i] = (__bf16)p[i];
#pragma unroll
  for (int i = 0; i < 8; ++i) a[8 + i] = (__bf16)p[8 + i];
  return a;
}

__device__ __forceinline__ v16bf load_a_bf16(const __bf16* p0, int ld, int k0, int lane) {
  int m = lane & 15, hi = lane >> 4;
  const __bf16* p = p0 + (size_t)m * ld + k0 + hi * 16;
  v16bf a;
#pragma unroll
  for (int i = 0; i < 8; ++i) a[i] = p[i];
#pragma unroll
  for (int i = 0; i < 8; ++i) a[8 + i] = p[8 + i];
  return a;
}

// B-fragment (32x16): lane n = lane&15, K half by lane>=16; B[k][n] = W[n][k]
// for W row-major with leading dim ld (i.e. we feed the transposed operand).
__device__ __forceinline__ v16bf load_b_f32(const float* p0, int ld, int k0, int lane) {
  int n = lane & 15, hi = lane >> 4;
  const float* p = p0 + (size_t)n * ld + k0 + hi * 16;
  v16bf b;
#pragma unroll
  for (int i = 0; i < 16; ++i) b[i] = (__bf16)p[i];
  return b;
}

__device__ __forceinline__ v16bf load_b_bf16(const __bf16* p0, int ld, int k0, int lane) {
  int n = lane & 15, hi = lane >> 4;
  const __bf16* p = p0 + (size_t)n * ld + k0 + hi * 16;
  v16bf b;
#pragma unroll
  for (int i = 0; i < 16; ++i) b[i] = p[i];
  return b;
}

// Sylvester-Hadamard B-fragment: H[k][n] = (-1)^popc(k&n), generated in regs.
__device__ __forceinline__ v16bf load_b_had(int k0, int n0, int lane) {
  int n = lane & 15, hi = lane >> 4;
  int ncol = n0 + n;
  v16bf b;
#pragma unroll
  for (int i = 0; i < 16; ++i) {
    int kk = k0 + hi * 16 + i;
    b[i] = (__bf16)((__popc(kk & ncol) & 1) ? -1.0f : 1.0f);
  }
  return b;
}

// ---------------------------------------------------------------------------
// TDM: issue a 2D bf16 tile load (rows x rowlen elems) global -> LDS.
// D# layout per CDNA5 ISA ch.8 (group0 128b, group1 256b; groups 2/3 zero for
// a 2D tensor). One issue per wave; completion via TENSORcnt.
// ---------------------------------------------------------------------------
__device__ __forceinline__ void tdm_load_2d_bf16(unsigned lds_off, const void* gaddr,
                                                 unsigned rows, unsigned rowlen,
                                                 unsigned tensor_d0, unsigned tensor_d1,
                                                 unsigned stride_elems) {
  unsigned long long ga = (unsigned long long)(uintptr_t)gaddr;
  v4u g0;
  g0[0] = 1u;                                            // count=1, user D#
  g0[1] = lds_off;                                       // lds_addr (bytes)
  g0[2] = (unsigned)(ga & 0xffffffffu);                  // global_addr[31:0]
  g0[3] = (unsigned)((ga >> 32) & 0x01ffffffu)           // global_addr[56:32]
          | 0x80000000u;                                 // type = 2 ("image")
  v8i g1;
  g1[0] = 0x00010000;                                    // data_size=1 (2 bytes)
  g1[1] = (int)((tensor_d0 & 0xffffu) << 16);            // tensor_dim0[15:0] @ b63:48
  g1[2] = (int)((tensor_d0 >> 16) | ((tensor_d1 & 0xffffu) << 16)); // td0 hi | td1 lo
  g1[3] = (int)(((tensor_d1 >> 16) & 0xffffu) | ((rowlen & 0xffffu) << 16)); // td1 hi | tile_dim0
  g1[4] = (int)(rows & 0xffffu);                         // tile_dim1 (tile_dim2=0)
  g1[5] = (int)stride_elems;                             // tensor_dim0_stride[31:0]
  g1[6] = 0;                                             // stride hi, dim1_stride
  g1[7] = 0;
  v4i gz = {0, 0, 0, 0};
#if defined(__clang_major__) && (__clang_major__ >= 23)
  v8i gz8 = {0, 0, 0, 0, 0, 0, 0, 0};
  __builtin_amdgcn_tensor_load_to_lds(g0, g1, gz, gz, gz8, 0);
#else
  __builtin_amdgcn_tensor_load_to_lds(g0, g1, gz, gz, 0);
#endif
}

// ---------------------------------------------------------------------------
// Kernel 1: w[g, h] = (x[g,:] . wproj[h,:]) * NH^-0.5   (g = b*S+s flattened)
// ---------------------------------------------------------------------------
__global__ void __launch_bounds__(64)
wproj_kernel(const float* __restrict__ x, const float* __restrict__ wp,
             float* __restrict__ w_out) {
  const int g0 = blockIdx.x * 16;
  const int lane = threadIdx.x & 31;
  const int wv = threadIdx.x >> 5;
  const int n0 = wv * 16;
  const float* xa = x + (size_t)g0 * DIMD;
  const float* wb = wp + (size_t)n0 * DIMD;
  v8f c = {};
  for (int k0 = 0; k0 < DIMD; k0 += 32) {
    v16bf a = load_a_f32(xa, DIMD, k0, lane);
    v16bf b = load_b_f32(wb, DIMD, k0, lane);
    c = wmma_bf16(a, b, c);
  }
  const int n = lane & 15, hi = lane >> 4;
#pragma unroll
  for (int r = 0; r < 8; ++r) {
    int m = r + hi * 8;
    w_out[(size_t)(g0 + m) * NH + n0 + n] = c[r] * NHSCALE;
  }
}

// ---------------------------------------------------------------------------
// Kernel 2: k-path. kf = x @ wk^T -> LayerNorm -> RoPE(first 64) ->
//           bf16 Hadamard (WMMA) * HD^-0.5 -> kk (bf16).
// ---------------------------------------------------------------------------
__global__ void __launch_bounds__(256)
knorm_kernel(const float* __restrict__ x,  const float* __restrict__ wk,
             const float* __restrict__ knw, const float* __restrict__ knb,
             const float* __restrict__ fcos, const float* __restrict__ fsin,
             __bf16* __restrict__ kk) {
  __shared__ float kf[16 * HD];
  __shared__ float psum[256], psq[256];
  __shared__ float mu_s[16], rstd_s[16];
  const int g0 = blockIdx.x * 16;
  const int tid = threadIdx.x;
  const int lane = tid & 31, wv = tid >> 5;
  const int n0 = wv * 16;

  { // GEMM: kf(16x128) = x_tile(16x4096) @ wk^T
    const float* xa = x + (size_t)g0 * DIMD;
    const float* wb = wk + (size_t)n0 * DIMD;
    v8f c = {};
    for (int k0 = 0; k0 < DIMD; k0 += 32) {
      v16bf a = load_a_f32(xa, DIMD, k0, lane);
      v16bf b = load_b_f32(wb, DIMD, k0, lane);
      c = wmma_bf16(a, b, c);
    }
    const int n = lane & 15, hi = lane >> 4;
#pragma unroll
    for (int r = 0; r < 8; ++r) kf[(r + hi * 8) * HD + n0 + n] = c[r];
  }
  __syncthreads();

  { // LayerNorm partials: 16 threads per row, 8 elems each
    int m = tid >> 4, seg = tid & 15;
    float s = 0.f, q = 0.f;
#pragma unroll
    for (int i = 0; i < 8; ++i) {
      float v = kf[m * HD + seg * 8 + i];
      s += v; q += v * v;
    }
    psum[tid] = s; psq[tid] = q;
  }
  __syncthreads();
  if (tid < 16) {
    float s = 0.f, q = 0.f;
    for (int i = 0; i < 16; ++i) { s += psum[tid * 16 + i]; q += psq[tid * 16 + i]; }
    float mu = s * (1.0f / HD);
    float var = q * (1.0f / HD) - mu * mu;
    mu_s[tid] = mu;
    rstd_s[tid] = rsqrtf(var + 1e-6f);
  }
  __syncthreads();
  for (int i = tid; i < 16 * HD; i += 256) {
    int m = i >> 7, d = i & 127;
    kf[i] = (kf[i] - mu_s[m]) * rstd_s[m] * knw[d] + knb[d];
  }
  __syncthreads();
  // RoPE on first RD=64 dims: pair (c, c+32)
  for (int p = tid; p < 16 * 32; p += 256) {
    int m = p >> 5, ci = p & 31;
    int s_loc = (g0 + m) % SS;
    float x1 = kf[m * HD + ci], x2 = kf[m * HD + 32 + ci];
    float cv = fcos[s_loc * 32 + ci], sv = fsin[s_loc * 32 + ci];
    kf[m * HD + ci]      = x1 * cv - x2 * sv;
    kf[m * HD + 32 + ci] = x1 * sv + x2 * cv;
  }
  __syncthreads();
  { // Hadamard via WMMA (B generated from popc signs), scale, store bf16
    v8f c = {};
#pragma unroll
    for (int k0 = 0; k0 < HD; k0 += 32) {
      v16bf a = load_a_f32(kf, HD, k0, lane);
      v16bf b = load_b_had(k0, n0, lane);
      c = wmma_bf16(a, b, c);
    }
    const int n = lane & 15, hi = lane >> 4;
#pragma unroll
    for (int r = 0; r < 8; ++r) {
      int m = r + hi * 8;
      kk[(size_t)(g0 + m) * HD + n0 + n] = (__bf16)(c[r] * HSCALE);
    }
  }
}

// ---------------------------------------------------------------------------
// Kernel 3: q-path with head reduction. The 16x1536 qr A-tile is staged once
// as bf16 in dynamic LDS (48 KB) and reused for all 32 heads.
// ---------------------------------------------------------------------------
__global__ void __launch_bounds__(256)
qhat_kernel(const float* __restrict__ qr, const float* __restrict__ wqb,
            const float* __restrict__ fcos, const float* __restrict__ fsin,
            const float* __restrict__ w_in, __bf16* __restrict__ qhat) {
  extern __shared__ __bf16 qrt[];                // 16*QLRK bf16 = 48 KB
  __shared__ float qh[16 * HD];
  __shared__ float acc[16 * HD];
  __shared__ float wrow[16];
  const int g0 = blockIdx.x * 16;
  const int tid = threadIdx.x;
  const int lane = tid & 31, wv = tid >> 5;
  const int n0 = wv * 16;

  const float* qa = qr + (size_t)g0 * QLRK;      // 16 contiguous rows
  for (int i = tid; i < 16 * QLRK; i += 256) qrt[i] = (__bf16)qa[i];
  for (int i = tid; i < 16 * HD; i += 256) acc[i] = 0.f;
  __syncthreads();

  for (int h = 0; h < NH; ++h) {
    if (tid < 16) wrow[tid] = w_in[(size_t)(g0 + tid) * NH + h];
    { // q head GEMM: (16 x 1536) @ (1536 x 128), A from LDS
      const float* wb = wqb + (size_t)(h * HD + n0) * QLRK;
      v8f c = {};
      for (int k0 = 0; k0 < QLRK; k0 += 32) {
        v16bf a = load_a_bf16(qrt, QLRK, k0, lane);
        v16bf b = load_b_f32(wb, QLRK, k0, lane);
        c = wmma_bf16(a, b, c);
      }
      const int n = lane & 15, hi = lane >> 4;
#pragma unroll
      for (int r = 0; r < 8; ++r) qh[(r + hi * 8) * HD + n0 + n] = c[r];
    }
    __syncthreads();
    // RoPE on first RD dims
    for (int p = tid; p < 16 * 32; p += 256) {
      int m = p >> 5, ci = p & 31;
      int s_loc = (g0 + m) % SS;
      float x1 = qh[m * HD + ci], x2 = qh[m * HD + 32 + ci];
      float cv = fcos[s_loc * 32 + ci], sv = fsin[s_loc * 32 + ci];
      qh[m * HD + ci]      = x1 * cv - x2 * sv;
      qh[m * HD + 32 + ci] = x1 * sv + x2 * cv;
    }
    __syncthreads();
    { // Hadamard-WMMA, weight by w[m,h], accumulate
      v8f c = {};
#pragma unroll
      for (int k0 = 0; k0 < HD; k0 += 32) {
        v16bf a = load_a_f32(qh, HD, k0, lane);
        v16bf b = load_b_had(k0, n0, lane);
        c = wmma_bf16(a, b, c);
      }
      const int n = lane & 15, hi = lane >> 4;
#pragma unroll
      for (int r = 0; r < 8; ++r) {
        int m = r + hi * 8;
        float qv = (float)(__bf16)(c[r] * HSCALE);            // ref: bf16 q_had
        float wq = (float)(__bf16)(qv * wrow[m] * SMSCALE);   // ref: bf16 wq
        acc[m * HD + n0 + n] += wq;                           // f32 head-sum
      }
    }
    __syncthreads();
  }
  for (int i = tid; i < 16 * HD; i += 256)
    qhat[(size_t)(g0 + (i >> 7)) * HD + (i & 127)] = (__bf16)acc[i];
}

// ---------------------------------------------------------------------------
// Kernel 4: k-tiles are DMA'd into LDS by the Tensor Data Mover (wave 0,
// TENSORcnt-waited), score stripe (16 x 4096) computed with WMMA into 256 KB
// of dynamic LDS, causal-masked, then per-row top-2048 via 4-pass radix select.
// ---------------------------------------------------------------------------
__global__ void __launch_bounds__(256)
score_topk_kernel(const __bf16* __restrict__ qhat, const __bf16* __restrict__ kk,
                  const int* __restrict__ sp_ptr, int* __restrict__ out) {
  extern __shared__ float scores[];              // [16][SS] = 256 KB (dynamic)
  __shared__ __bf16 ktile[128 * HD];             // 32 KB TDM staging tile
  __shared__ unsigned hist[8][256];
  __shared__ unsigned bc[8][2];
  __shared__ unsigned cnt[8][2];
  const int g0 = blockIdx.x * 16;
  const int batch = g0 / SS;
  const int s0 = g0 % SS;
  const int tid = threadIdx.x;
  const int lane = tid & 31, wv = tid >> 5;
  const int sp = *sp_ptr;

  const __bf16* qa = qhat + (size_t)g0 * HD;
  const __bf16* kb = kk + (size_t)batch * SS * HD;
  const unsigned ktile_lds = (unsigned)(uintptr_t)(void*)ktile;

  // Hoist A fragments (reused for every t-tile)
  v16bf afr[4];
#pragma unroll
  for (int kc = 0; kc < 4; ++kc) afr[kc] = load_a_bf16(qa, HD, kc * 32, lane);

  for (int t0 = 0; t0 < SS; t0 += 128) {
    if (wv == 0) {                               // one TDM issue per tile
      tdm_load_2d_bf16(ktile_lds, kb + (size_t)t0 * HD,
                       /*rows=*/128, /*rowlen=*/HD,
                       /*tensor_d0=*/HD, /*tensor_d1=*/SS, /*stride=*/HD);
      __builtin_amdgcn_s_wait_tensorcnt(0);      // TENSORcnt is per-wave
    }
    __syncthreads();                             // tile visible to all waves

    const int tt = wv * 16;                      // local row base in ktile
    v8f c = {};
#pragma unroll
    for (int kc = 0; kc < 4; ++kc) {
      v16bf b = load_b_bf16(ktile + (size_t)tt * HD, HD, kc * 32, lane);
      c = wmma_bf16(afr[kc], b, c);
    }
    const int n = lane & 15, hi = lane >> 4;
#pragma unroll
    for (int r = 0; r < 8; ++r) {
      int m = r + hi * 8;
      int t = t0 + tt + n;
      float v = c[r];
      if (t > s0 + m + sp) v += -1e30f;          // causal mask (col <= row)
      scores[m * SS + t] = v;
    }
    __syncthreads();                             // before TDM overwrites tile
  }

  // Per-row top-K: wave wv handles rows wv and wv+8 (uniform barrier counts).
  for (int rr = 0; rr < 2; ++rr) {
    const int m = wv + rr * 8;
    const float* row = scores + (size_t)m * SS;
    unsigned prefix = 0;
    int remaining = TOPKN;
    for (int p = 3; p >= 0; --p) {
      for (int i = lane; i < 256; i += 32) hist[wv][i] = 0;
      __syncthreads();
      for (int t = lane; t < SS; t += 32) {
        unsigned u = __float_as_uint(row[t]);
        u = (u & 0x80000000u) ? ~u : (u | 0x80000000u);   // order-preserving key
        bool match = (p == 3) || ((u >> (8 * (p + 1))) == prefix);
        if (match) atomicAdd(&hist[wv][(u >> (8 * p)) & 255u], 1u);
      }
      __syncthreads();
      if (lane == 0) {
        unsigned cum = 0, dsel = 0, above = 0;
        for (int d = 255; d >= 0; --d) {
          unsigned hc = hist[wv][d];
          if (cum + hc >= (unsigned)remaining) { dsel = (unsigned)d; above = cum; break; }
          cum += hc;
        }
        bc[wv][0] = dsel; bc[wv][1] = above;
      }
      __syncthreads();
      prefix = (prefix << 8) | bc[wv][0];
      remaining -= (int)bc[wv][1];
    }
    const unsigned thresh = prefix;          // key of the TOPK-th largest
    const int tie_base = TOPKN - remaining;  // #elements strictly greater
    if (lane == 0) { cnt[wv][0] = 0; cnt[wv][1] = 0; }
    __syncthreads();
    int* orow = out + (size_t)(g0 + m) * TOPKN;
    for (int t = lane; t < SS; t += 32) {
      unsigned u = __float_as_uint(row[t]);
      u = (u & 0x80000000u) ? ~u : (u | 0x80000000u);
      if (u > thresh) {
        unsigned pos = atomicAdd(&cnt[wv][0], 1u);
        orow[pos] = t;
      } else if (u == thresh) {
        unsigned pos = (unsigned)tie_base + atomicAdd(&cnt[wv][1], 1u);
        if (pos < TOPKN) orow[pos] = t;
      }
    }
    __syncthreads();
  }
}

// ---------------------------------------------------------------------------
extern "C" void kernel_launch(void* const* d_in, const int* in_sizes, int n_in,
                              void* d_out, int out_size, void* d_ws, size_t ws_size,
                              hipStream_t stream) {
  (void)in_sizes; (void)n_in; (void)out_size; (void)ws_size;
  const float* x    = (const float*)d_in[0];
  const float* qr   = (const float*)d_in[1];
  const float* wq_b = (const float*)d_in[2];
  const float* wk_w = (const float*)d_in[3];
  const float* knw  = (const float*)d_in[4];
  const float* knb  = (const float*)d_in[5];
  const float* wpj  = (const float*)d_in[6];
  const float* fc   = (const float*)d_in[7];
  const float* fs   = (const float*)d_in[8];
  // d_in[9] = mask (recomputed on device), d_in[10] = start_pos scalar
  const int* sp     = (const int*)d_in[10];
  int* out = (int*)d_out;

  char* ws = (char*)d_ws;
  float*  w_f32 = (float*)ws;                                   // 1 MB
  __bf16* qhat  = (__bf16*)(ws + (size_t)BS * NH * 4);          // 2 MB
  __bf16* kkb   = (__bf16*)(ws + (size_t)BS * NH * 4 + (size_t)BS * HD * 2); // 2 MB

  const dim3 grd(BS / 16);
  wproj_kernel<<<grd, dim3(64), 0, stream>>>(x, wpj, w_f32);
  knorm_kernel<<<grd, dim3(256), 0, stream>>>(x, wk_w, knw, knb, fc, fs, kkb);

  const int qr_lds = 16 * QLRK * (int)sizeof(__bf16);           // 48 KB
  hipFuncSetAttribute(reinterpret_cast<const void*>(qhat_kernel),
                      hipFuncAttributeMaxDynamicSharedMemorySize, qr_lds);
  qhat_kernel<<<grd, dim3(256), qr_lds, stream>>>(qr, wq_b, fc, fs, w_f32, qhat);

  const int score_lds = 16 * SS * (int)sizeof(float);           // 256 KB
  hipFuncSetAttribute(reinterpret_cast<const void*>(score_topk_kernel),
                      hipFuncAttributeMaxDynamicSharedMemorySize, score_lds);
  score_topk_kernel<<<grd, dim3(256), score_lds, stream>>>(qhat, kkb, sp, out);
}
